// QuantLinear_44702019617444
// MI455X (gfx1250) — matrix operand order repaired and checked by reading, not compiled, not run on hardware
//
#include <hip/hip_runtime.h>
#include <stdint.h>

typedef __attribute__((ext_vector_type(16))) _Float16 v16h;
typedef __attribute__((ext_vector_type(8)))  _Float16 v8h;
typedef __attribute__((ext_vector_type(8)))  float    v8f;
typedef __attribute__((ext_vector_type(4)))  int      v4i;
typedef __attribute__((ext_vector_type(8)))  unsigned v8u;
typedef __attribute__((ext_vector_type(4)))  unsigned u32x4;
typedef __attribute__((ext_vector_type(8)))  int      i32x8;
typedef __attribute__((ext_vector_type(4)))  int      i32x4;

#define K_DIM   4096
#define N_DIM   11008
#define M_DIM   32
#define KSTEP   128                 // K per iteration (4 WMMAs / wave)
#define KITERS  (K_DIM / KSTEP)     // 32 iterations

#if __has_builtin(__builtin_amdgcn_tensor_load_to_lds)
#define HAVE_TDM 1
#endif

// ---------------------------------------------------------------------------
// Prep: x (f32) -> f16 workspace copy + per-row sums (for the affine epilogue
// y = s*acc - (1024*s + z)*rowsum + bias, where acc = x_f16 @ (q + 1024)).
// ---------------------------------------------------------------------------
__global__ __launch_bounds__(256) void xprep_kernel(const float* __restrict__ x,
                                                    _Float16* __restrict__ xh,
                                                    float* __restrict__ rowsum) {
  __shared__ float red[256];
  const int row = blockIdx.x;     // 0..31
  const int t   = threadIdx.x;    // 0..255
  float s = 0.0f;
  const float* xr = x + row * K_DIM;
  _Float16* xo = xh + row * K_DIM;
  for (int i = t; i < K_DIM; i += 256) {
    float v = xr[i];
    xo[i] = (_Float16)v;
    s += v;
  }
  red[t] = s;
  __syncthreads();
  for (int off = 128; off > 0; off >>= 1) {
    if (t < off) red[t] += red[t + off];
    __syncthreads();
  }
  if (t == 0) rowsum[row] = red[0];
}

#ifdef HAVE_TDM
// Issue one TDM descriptor: load a 16-row x 32-col int32 tile (row stride
// N_DIM words) from the packed-weight tensor into contiguous LDS.
// D# layout per CDNA5 ISA ch.8 (group0: count/lds/global/type; group1:
// data_size, tensor dims, tile dims, stride; groups 2/3 zero for 2-D).
static __device__ __forceinline__ void tdm_load_tile(uint64_t gaddr, unsigned lds_addr) {
  u32x4 g0;
  g0[0] = 1u;                                       // count=1 (valid user D#)
  g0[1] = lds_addr;                                 // LDS byte address
  g0[2] = (unsigned)gaddr;                          // global_addr[31:0]
  g0[3] = (unsigned)(gaddr >> 32) | 0x80000000u;    // global_addr[56:32] | type=2
  i32x8 g1;
  g1[0] = 0x00020000;                               // data_size = 4 bytes
  g1[1] = (int)((unsigned)N_DIM << 16);             // tensor_dim0[15:0] @ bits63:48
  g1[2] = (int)((unsigned)(K_DIM / 8) << 16);       // tensor_dim1[15:0] @ bits95:80
  g1[3] = (int)(32u << 16);                         // tile_dim0 = 32   @ bits127:112
  g1[4] = 16;                                       // tile_dim1 = 16   @ bits143:128
  g1[5] = N_DIM;                                    // tensor_dim0_stride[31:0]
  g1[6] = 0;
  g1[7] = 0;
  i32x4 z4 = {};
#if __clang_major__ >= 23
  i32x8 z8 = {};
  __builtin_amdgcn_tensor_load_to_lds(g0, g1, z4, z4, z8, 0);
#else
  __builtin_amdgcn_tensor_load_to_lds(g0, g1, z4, z4, 0);
#endif
}
#endif

// ---------------------------------------------------------------------------
// Main GEMM. Block: 128 threads = 4 waves (2 M-tiles x 2 N-tiles of 16).
// Grid: N_DIM/32 = 344 blocks.
// Double-buffered pipeline: wave 0 issues a single TDM tensor_load_to_lds for
// chunk kc+1 (TENSORcnt) before the block computes 4 WMMAs from chunk kc,
// then waits tensorcnt + one barrier. The DMA engine does all the address
// generation and overlaps the tensor-core work.
// Nibbles -> f16 via the 0x6400 exponent trick (value = 1024 + q, folded into
// the epilogue), so dequant is pure bfe/or3, zero float converts.
// ---------------------------------------------------------------------------
__global__ __launch_bounds__(128) void qmm_kernel(const _Float16* __restrict__ xh,
                                                  const int* __restrict__ qw,
                                                  const float* __restrict__ scales,
                                                  const float* __restrict__ zeros,
                                                  const float* __restrict__ bias,
                                                  const float* __restrict__ rowsum,
                                                  float* __restrict__ out) {
  __shared__ int lq[2][512];             // 2 buffers x (16 K-word rows x 32 N cols)

  const int t    = threadIdx.x;
  const int lane = t & 31;
  const int w    = t >> 5;               // wave 0..3
  const int mt   = w & 1;                // M tile (0,1)
  const int nt   = w >> 1;               // N tile within block (0,1)
  const int n0   = blockIdx.x * 32;
  const int nw   = n0 + nt * 16;         // this wave's N base
  const int lrow = lane & 15;
  const int lhi  = lane >> 4;

  // A-fragment addressing (16-bit A 16x32 layout):
  //   lanes 0-15 : row M=lane,    K = 0..7  and 16..23
  //   lanes 16-31: row M=lane-16, K = 8..15 and 24..31
  const int arow  = mt * 16 + lrow;
  const int akoff = lhi ? 8 : 0;
  const _Float16* aptr = xh + arow * K_DIM + akoff;

  // B-fragment addressing (16-bit B 32x16 layout):
  //   lane holds column N = lane%16; lanes 0-15 K = 0..7, 16..23; lanes 16-31 K = 8..15, 24..31.
  const int krLocal   = lrow + 16 * lhi;     // K row within 32 sub-chunk
  const int qrowLocal = krLocal >> 3;        // packed-word row 0..3
  const int shamt     = (krLocal & 7) * 4;   // nibble shift inside the word

  const unsigned lbuf0 = (unsigned)(uintptr_t)&lq[0][0];

#ifndef HAVE_TDM
  // Fallback staging: 512 words / iteration, 4 async b32 loads per thread.
  const int srow = t >> 5;
  const int scol = t & 31;
  const int* gbase = qw + (size_t)srow * N_DIM + n0 + scol;
  const unsigned lbase = lbuf0 + (unsigned)(t * 4);
#endif

  auto stage = [&](int kc, int buf) {
#ifdef HAVE_TDM
    if (w == 0) {
      uint64_t g = (uint64_t)(const void*)(qw + (size_t)kc * 16 * N_DIM + n0);
      tdm_load_tile(g, lbuf0 + (unsigned)(buf * 2048));
    }
#else
#pragma unroll
    for (int i = 0; i < 4; ++i) {
      uint64_t g = (uint64_t)(const void*)(gbase + (size_t)(kc * 16 + 4 * i) * N_DIM);
      unsigned  l = lbase + (unsigned)(buf * 2048 + i * 512);
      asm volatile("global_load_async_to_lds_b32 %0, %1, off"
                   :: "v"(l), "v"(g) : "memory");
    }
#endif
  };

  auto stage_wait = [&]() {
#ifdef HAVE_TDM
    if (w == 0) __builtin_amdgcn_s_wait_tensorcnt(0);
#else
    asm volatile("s_wait_asynccnt 0x0" ::: "memory");
#endif
    __syncthreads();
  };

  v8f acc = {};

  // Prologue: fill buffer 0.
  stage(0, 0);
  stage_wait();

  for (int kc = 0; kc < KITERS; ++kc) {
    const int cur = kc & 1;
    // Kick off DMA for the next chunk before computing this one.
    if (kc + 1 < KITERS) stage(kc + 1, cur ^ 1);

#pragma unroll
    for (int c = 0; c < 4; ++c) {
      // ---- A fragment: 16 f16 per lane from the f16 copy of x ----
      const _Float16* ap = aptr + kc * KSTEP + c * 32;
      v8h alo = *(const v8h*)(ap);
      v8h ahi = *(const v8h*)(ap + 16);
      v16h a;
#pragma unroll
      for (int i = 0; i < 8; ++i) { a[i] = alo[i]; a[8 + i] = ahi[i]; }

      // ---- B fragment: this lane's column, K = 8*lhi+0..7 and 16+8*lhi+0..7 ----
      const int* lp = &lq[cur][(c * 4 + lhi) * 32 + nt * 16 + lrow];
      const unsigned wlo = (unsigned)lp[0], whi = (unsigned)lp[64];
      // f16 bit trick: 0x6400 | q  ==  (f16)(1024 + q), exact for q in [0,15]
      v8u bb;
#pragma unroll
      for (int j = 0; j < 8; ++j) {
        const unsigned wd = (j < 4) ? wlo : whi;
        const int sh = (j & 3) * 8;
        unsigned lo = (wd >> sh) & 0xFu;
        unsigned hi = (wd >> (sh + 4)) & 0xFu;
        bb[j] = 0x64006400u | lo | (hi << 16);
      }
      v16h b = __builtin_bit_cast(v16h, bb);

      acc = __builtin_amdgcn_wmma_f32_16x16x32_f16(
          /*neg_a=*/false, a, /*neg_b=*/false, b,
          /*c_mod=*/(short)0, acc, /*reuse_a=*/false, /*reuse_b=*/false);
    }

    // DMA into the other buffer must land before anyone reads it.
    stage_wait();
  }

  // ---- epilogue: y = s*acc - (1024*s + z)*rowsum + bias ----
  // C/D layout: lanes 0-15 -> M = r, N = lane; lanes 16-31 -> M = 8 + r.
  const int n  = nw + lrow;
  const float sn = scales[n];
  const float zf = fmaf(1024.0f, sn, zeros[n]);
  const float bn = bias[n];
#pragma unroll
  for (int r = 0; r < 8; ++r) {
    const int m = mt * 16 + 8 * lhi + r;
    out[m * N_DIM + n] = sn * acc[r] - zf * rowsum[m] + bn;
  }
}

// ---------------------------------------------------------------------------
extern "C" void kernel_launch(void* const* d_in, const int* in_sizes, int n_in,
                              void* d_out, int out_size, void* d_ws, size_t ws_size,
                              hipStream_t stream) {
  const float* x      = (const float*)d_in[0];   // [32,4096]
  const int*   qw     = (const int*)d_in[1];     // [512,11008]
  const float* scales = (const float*)d_in[2];   // [11008]
  const float* zeros  = (const float*)d_in[3];   // [11008]
  const float* bias   = (const float*)d_in[4];   // [11008]
  float* out = (float*)d_out;                    // [32,11008]

  _Float16* xh  = (_Float16*)d_ws;                                   // 256 KB
  float* rowsum = (float*)((char*)d_ws + (size_t)M_DIM * K_DIM * 2); // 128 B

  xprep_kernel<<<M_DIM, 256, 0, stream>>>(x, xh, rowsum);
  qmm_kernel<<<N_DIM / 32, 128, 0, stream>>>(xh, qw, scales, zeros, bias, rowsum, out);
}
